// SimpleMoEExperts_558345748976
// MI455X (gfx1250) — compile-verified, matrix operand order checked by reading, not run on hardware
//
#include <hip/hip_runtime.h>
#include <hip/hip_bf16.h>

typedef __attribute__((ext_vector_type(4)))  __bf16 v4bf;
typedef __attribute__((ext_vector_type(8)))  __bf16 v8bf;
typedef __attribute__((ext_vector_type(16))) __bf16 v16bf;
typedef __attribute__((ext_vector_type(8)))  float  v8f;

static constexpr int E_ = 8;
static constexpr int H_ = 2048;
static constexpr int I_ = 1408;
static constexpr int T_ = 16384;
static constexpr int BK   = 32;   // K tile (bf16 WMMA K=32)
static constexpr int LDSK = 40;   // padded LDS row stride in bf16 (80B, keeps 16B chunks aligned)

// CDNA5 async global->LDS path (ASYNCcnt-tracked). The clang builtin exists but its
// pointer params live in LangAS::cuda_device/cuda_shared, which user code cannot
// spell — so use inline asm (the ISA-portable path). Device-only: host pass takes
// the plain-copy fallback.
#if defined(__gfx1250__)
#define USE_ASYNC_LDS 1
#else
#define USE_ASYNC_LDS 0
#endif

#if USE_ASYNC_LDS
__device__ __forceinline__ void async_g2l_b128(const void* g, void* l) {
  const unsigned long long ga = (unsigned long long)(uintptr_t)g;
  const unsigned la = (unsigned)(uintptr_t)l;   // generic LDS addr: low 32 bits = LDS offset
  asm volatile("global_load_async_to_lds_b128 %0, %1, off"
               :: "v"(la), "v"(ga) : "memory");
}
__device__ __forceinline__ void wait_async0() {
  asm volatile("s_wait_asynccnt 0" ::: "memory");
}
#endif

__device__ __forceinline__ int expert_of(const int* __restrict__ counts, int token) {
  int off = 0;
#pragma unroll
  for (int e = 0; e < E_; ++e) {
    int c = counts[e];
    if (token < off + c) return e;
    off += c;
  }
  return E_ - 1;
}

// 16x32 bf16 WMMA operand fragment from an LDS tile.
// Lane l: m = l%16, h = l/16 -> K elements [8h,8h+8) ++ [16+8h,24+8h)
__device__ __forceinline__ v16bf frag16(const __bf16* __restrict__ s, int row, int k0) {
  const v8bf lo = *(const v8bf*)(s + row * LDSK + k0);
  const v8bf hi = *(const v8bf*)(s + row * LDSK + k0 + 16);
  return __builtin_shufflevector(lo, hi, 0,1,2,3,4,5,6,7,8,9,10,11,12,13,14,15);
}

__device__ __forceinline__ v4bf cvt4(const float4 v) {
  v4bf b;
  b[0] = (__bf16)v.x; b[1] = (__bf16)v.y; b[2] = (__bf16)v.z; b[3] = (__bf16)v.w;
  return b;
}

// ---------------- GEMM1: gate_up = x @ Wgu^T, h = silu(gate)*up -> bf16 ws ----------------
// Block tile: 128(M tokens) x 64(N over I) x K=32. 8 waves: 4 over M, 2 over N.
// Double-buffered LDS pipeline: global->regs for tile k+1 issued before computing tile k.
__global__ __launch_bounds__(256)
void moe_gemm1_silu(const float* __restrict__ x,
                    const float* __restrict__ gup,   // [E, 2I, H] row-major
                    const int* __restrict__ counts,
                    __bf16* __restrict__ hbuf) {     // [T, I] bf16
  __shared__ __align__(16) __bf16 sX [2][128 * LDSK];
  __shared__ __align__(16) __bf16 sWg[2][ 64 * LDSK];
  __shared__ __align__(16) __bf16 sWu[2][ 64 * LDSK];

  const int tBase = blockIdx.x * 128;
  const int iBase = blockIdx.y * 64;
  const int e = expert_of(counts, tBase);

  const int tid  = threadIdx.x;
  const int lane = tid & 31;
  const int wave = tid >> 5;
  const int wm = wave & 3;   // wave row:  32 tokens
  const int wn = wave >> 2;  // wave col:  32 i-values
  const int lm  = lane & 15;
  const int k0l = (lane >> 4) * 8;
  const int lh8 = (lane >> 4) * 8;

  const float* __restrict__ xBase  = x   + (size_t)tBase * H_;
  const float* __restrict__ wgBase = gup + ((size_t)e * 2 * I_ + iBase)      * H_;
  const float* __restrict__ wuBase = gup + ((size_t)e * 2 * I_ + I_ + iBase) * H_;

  // staging geometry: thread -> (row = tid/8 + p*32, col = (tid%8)*4)
  const int srow = tid >> 3;
  const int scol = (tid & 7) * 4;

  float4 rx[4], rg[2], ru[2];

  auto g2r = [&](int k) {
#pragma unroll
    for (int p = 0; p < 4; ++p)
      rx[p] = *(const float4*)(xBase + (size_t)(srow + p * 32) * H_ + k + scol);
#pragma unroll
    for (int p = 0; p < 2; ++p) {
      rg[p] = *(const float4*)(wgBase + (size_t)(srow + p * 32) * H_ + k + scol);
      ru[p] = *(const float4*)(wuBase + (size_t)(srow + p * 32) * H_ + k + scol);
    }
  };
  auto r2s = [&](int b) {
#pragma unroll
    for (int p = 0; p < 4; ++p)
      *(v4bf*)(sX[b] + (srow + p * 32) * LDSK + scol) = cvt4(rx[p]);
#pragma unroll
    for (int p = 0; p < 2; ++p) {
      *(v4bf*)(sWg[b] + (srow + p * 32) * LDSK + scol) = cvt4(rg[p]);
      *(v4bf*)(sWu[b] + (srow + p * 32) * LDSK + scol) = cvt4(ru[p]);
    }
  };

  v8f accg[2][2] = {};
  v8f accu[2][2] = {};

  g2r(0);
  r2s(0);
  constexpr int KT = H_ / BK;   // 64
  for (int kt = 0; kt < KT; ++kt) {
    __syncthreads();
    if (kt + 1 < KT) g2r((kt + 1) * BK);

    const int b = kt & 1;
    v16bf fa[2], fg[2], fu[2];
#pragma unroll
    for (int mi = 0; mi < 2; ++mi) fa[mi] = frag16(sX[b], wm * 32 + mi * 16 + lm, k0l);
#pragma unroll
    for (int nj = 0; nj < 2; ++nj) {
      fg[nj] = frag16(sWg[b], wn * 32 + nj * 16 + lm, k0l);
      fu[nj] = frag16(sWu[b], wn * 32 + nj * 16 + lm, k0l);
    }
#pragma unroll
    for (int mi = 0; mi < 2; ++mi)
#pragma unroll
      for (int nj = 0; nj < 2; ++nj) {
        accg[mi][nj] = __builtin_amdgcn_wmma_f32_16x16x32_bf16(
            false, fa[mi], false, fg[nj], (short)0, accg[mi][nj], false, false);
        accu[mi][nj] = __builtin_amdgcn_wmma_f32_16x16x32_bf16(
            false, fa[mi], false, fu[nj], (short)0, accu[mi][nj], false, false);
      }
    if (kt + 1 < KT) r2s((kt + 1) & 1);
  }

  // epilogue: h = silu(gate) * up   (fast rcp instead of IEEE divide; ref math is bf16 anyway)
#pragma unroll
  for (int mi = 0; mi < 2; ++mi) {
#pragma unroll
    for (int nj = 0; nj < 2; ++nj) {
      const int col  = iBase + wn * 32 + nj * 16 + lm;
      const int rowb = tBase + wm * 32 + mi * 16 + lh8;
#pragma unroll
      for (int r = 0; r < 8; ++r) {
        const float g = accg[mi][nj][r];
        const float u = accu[mi][nj][r];
        const float s = g * __builtin_amdgcn_rcpf(1.0f + __expf(-g));
        hbuf[(size_t)(rowb + r) * I_ + col] = (__bf16)(s * u);
      }
    }
  }
}

// ---------------- GEMM2: out = h @ Wdn^T -> f32 ----------------
// Block tile: 128(M) x 128(N over H) x K=32. 8 waves: 2 over M (64 rows), 4 over N (32 cols).
// h tile is raw bf16 -> async global->LDS copy; weight tile converts f32->bf16 through VGPRs.
__global__ __launch_bounds__(256)
void moe_gemm2(const __bf16* __restrict__ hbuf,     // [T, I] bf16
               const float* __restrict__ dwn,       // [E, H, I] row-major
               const int* __restrict__ counts,
               float* __restrict__ out) {           // [T, H] f32
  __shared__ __align__(16) __bf16 sH[2][128 * LDSK];
  __shared__ __align__(16) __bf16 sW[2][128 * LDSK];

  const int tBase = blockIdx.x * 128;
  const int nBase = blockIdx.y * 128;
  const int e = expert_of(counts, tBase);

  const int tid  = threadIdx.x;
  const int lane = tid & 31;
  const int wave = tid >> 5;
  const int wm = wave & 1;   // 64 rows
  const int wn = wave >> 1;  // 32 cols
  const int lm  = lane & 15;
  const int k0l = (lane >> 4) * 8;
  const int lh8 = (lane >> 4) * 8;

  const __bf16* __restrict__ hBase = hbuf + (size_t)tBase * I_;
  const float*  __restrict__ wBase = dwn + ((size_t)e * H_ + nBase) * I_;

  // staging geometry
  const int hrow = tid >> 2;          // + p*64 ; chunk = (tid%4)*8 bf16 (16B)
  const int hch  = (tid & 3) * 8;
  const int wrow = tid >> 3;          // + p*32 ; col = (tid%8)*4 f32
  const int wcol = (tid & 7) * 4;

  float4 rw[4];
#if !USE_ASYNC_LDS
  uint4 rh[2];
#endif

  auto g2r = [&](int k, int bnext) {
#if USE_ASYNC_LDS
#pragma unroll
    for (int p = 0; p < 2; ++p) {
      const __bf16* g = hBase + (size_t)(hrow + p * 64) * I_ + k + hch;
      __bf16* l = sH[bnext] + (hrow + p * 64) * LDSK + hch;
      async_g2l_b128(g, l);
    }
#else
    (void)bnext;
#pragma unroll
    for (int p = 0; p < 2; ++p)
      rh[p] = *(const uint4*)(hBase + (size_t)(hrow + p * 64) * I_ + k + hch);
#endif
#pragma unroll
    for (int p = 0; p < 4; ++p)
      rw[p] = *(const float4*)(wBase + (size_t)(wrow + p * 32) * I_ + k + wcol);
  };
  auto r2s = [&](int b) {
#if !USE_ASYNC_LDS
#pragma unroll
    for (int p = 0; p < 2; ++p)
      *(uint4*)(sH[b] + (hrow + p * 64) * LDSK + hch) = rh[p];
#endif
#pragma unroll
    for (int p = 0; p < 4; ++p)
      *(v4bf*)(sW[b] + (wrow + p * 32) * LDSK + wcol) = cvt4(rw[p]);
  };

  v8f acc[4][2] = {};

  g2r(0, 0);
  r2s(0);
  constexpr int KT = I_ / BK;   // 44
  for (int kt = 0; kt < KT; ++kt) {
#if USE_ASYNC_LDS
    wait_async0();              // publish async writes into sH[kt&1] before the barrier
#endif
    __syncthreads();
    if (kt + 1 < KT) g2r((kt + 1) * BK, (kt + 1) & 1);

    const int b = kt & 1;
    v16bf fa[4], fb[2];
#pragma unroll
    for (int mi = 0; mi < 4; ++mi) fa[mi] = frag16(sH[b], wm * 64 + mi * 16 + lm, k0l);
#pragma unroll
    for (int nj = 0; nj < 2; ++nj) fb[nj] = frag16(sW[b], wn * 32 + nj * 16 + lm, k0l);
#pragma unroll
    for (int mi = 0; mi < 4; ++mi)
#pragma unroll
      for (int nj = 0; nj < 2; ++nj)
        acc[mi][nj] = __builtin_amdgcn_wmma_f32_16x16x32_bf16(
            false, fa[mi], false, fb[nj], (short)0, acc[mi][nj], false, false);
    if (kt + 1 < KT) r2s((kt + 1) & 1);
  }

#pragma unroll
  for (int mi = 0; mi < 4; ++mi) {
#pragma unroll
    for (int nj = 0; nj < 2; ++nj) {
      const int col  = nBase + wn * 32 + nj * 16 + lm;
      const int rowb = tBase + wm * 64 + mi * 16 + lh8;
#pragma unroll
      for (int r = 0; r < 8; ++r)
        out[(size_t)(rowb + r) * H_ + col] = acc[mi][nj][r];
    }
  }
}

extern "C" void kernel_launch(void* const* d_in, const int* in_sizes, int n_in,
                              void* d_out, int out_size, void* d_ws, size_t ws_size,
                              hipStream_t stream) {
  const float* x      = (const float*)d_in[0];   // [T, H]
  const float* gup    = (const float*)d_in[1];   // [E, 2I, H]
  const float* dwn    = (const float*)d_in[2];   // [E, H, I]
  const int*   counts = (const int*)d_in[3];     // [E]
  float* out = (float*)d_out;                    // [T, H] f32
  __bf16* hbuf = (__bf16*)d_ws;                  // [T, I] bf16 staging (~46 MB)

  dim3 g1(T_ / 128, I_ / 64);    // 128 x 22
  moe_gemm1_silu<<<g1, dim3(256), 0, stream>>>(x, gup, counts, hbuf);

  dim3 g2(T_ / 128, H_ / 128);   // 128 x 16
  moe_gemm2<<<g2, dim3(256), 0, stream>>>(hbuf, dwn, counts, out);
}
